// GCN_Link_28346784154172
// MI455X (gfx1250) — matrix operand order repaired and checked by reading, not compiled, not run on hardware
//
#include <hip/hip_runtime.h>

// ---------------------------------------------------------------------------
// GCN link prediction forward, MI455X (gfx1250), wave32 + WMMA bf16.
//   T1t = (x @ W1)^T           (bf16)
//   H   = relu(adj @ T1 + b1)  (bf16)
//   T2t = (H @ W2)^T           (bf16)
//   Z   = adj @ T2 + b2        (bf16)
//   out = sigmoid(Z @ Z^T)     (f32, nontemporal) -> d_out
// Generic GEMM: C[M,N] = A[M,K] * Bt[N,K]^T, bf16 in / f32 accum via
// v_wmma_f32_16x16x32_bf16; async global->LDS double-buffered pipeline
// (global_load_async_to_lds_b128 + s_wait_asynccnt); BK=64 stages so each
// barrier pair covers 16 WMMAs per wave (TN=4).
// ---------------------------------------------------------------------------

typedef __attribute__((ext_vector_type(16))) __bf16 v16bf;
typedef __attribute__((ext_vector_type(8)))  float  v8f;
typedef __attribute__((ext_vector_type(4)))  float  f32x4;
typedef __attribute__((ext_vector_type(4)))  unsigned short u16x4;

#define BM 128
#define BKS 64                 // K depth per pipeline stage (2 WMMA K-steps)

__device__ __forceinline__ unsigned short f2bf_bits(float f) {
  __bf16 b = (__bf16)f;
  return __builtin_bit_cast(unsigned short, b);
}

// Async DMA: global (saddr + 32-bit voff) -> LDS (lds byte offset), 16 B/lane.
// Tracked by ASYNCcnt (CDNA5 ISA).
__device__ __forceinline__ void async_ld_b128(unsigned int lds_off,
                                              unsigned long long gbase,
                                              unsigned int goff) {
  asm volatile("global_load_async_to_lds_b128 %0, %1, %2"
               :: "v"(lds_off), "v"(goff), "s"(gbase)
               : "memory");
}

// A fragment (16x32 bf16, ISA §7.12.2 "16-bit A-Matrix 16x32"):
// lanes 0-15 hold M=0..15; half=lane/16 selects K groups {0..7,16..23} vs
// {8..15,24..31}; K-contiguous LDS rows -> two ds_load_b128.
__device__ __forceinline__ v16bf load_frag_a(const unsigned int* rowp, int hf) {
  union { v16bf v; uint4 q[2]; } u;
  u.q[0] = *(const uint4*)(rowp + hf * 4);
  u.q[1] = *(const uint4*)(rowp + hf * 4 + 8);
  return u.v;
}

// B fragment (32x16 bf16): same K order as A: half selects K groups
// {0..7,16..23} vs {8..15,24..31} -> two b128 loads.
__device__ __forceinline__ v16bf load_frag_b(const unsigned int* rowp, int hf) {
  union { v16bf v; uint4 q[2]; } u;
  u.q[0] = *(const uint4*)(rowp + hf * 4);
  u.q[1] = *(const uint4*)(rowp + hf * 4 + 8);
  return u.v;
}

#define WMMA_BF16(a, b, c) \
  __builtin_amdgcn_wmma_f32_16x16x32_bf16(false, (a), false, (b), (short)0, (c), false, false)

// TN = 16x16 N-tiles per wave (2 -> BN=64, 4 -> BN=128). Wave grid 4(M) x 2(N),
// wave tile = 32 x (16*TN). flags: bit0 = relu, bit1 = sigmoid.
template <int TN>
__global__ void __launch_bounds__(256)
gcn_wmma_gemm_bf16(const __bf16* __restrict__ A, int lda,
                   const __bf16* __restrict__ Bt, int ldb,
                   const float* __restrict__ bias,
                   float* __restrict__ outF, int ldf,
                   __bf16* __restrict__ outB, int ldob,
                   __bf16* __restrict__ outT, int ldt,
                   int K, int flags) {
  constexpr int BN     = 32 * TN;            // 64 or 128
  constexpr int ASZb   = BM * BKS * 2;       // 16 KB A stage
  constexpr int BSZb   = BN * BKS * 2;       // 8/16 KB B stage
  constexpr int STAGEb = ASZb + BSZb;
  constexpr int NCPYA  = ASZb / 4096;        // async instrs per thread, A (4)
  constexpr int NCPYB  = BSZb / 4096;        // async instrs per thread, B (2/4)
  constexpr int NASYNC = NCPYA + NCPYB;      // per-stage ASYNCcnt per wave

  __shared__ __align__(16) unsigned char smem[2 * STAGEb];

  const int t    = threadIdx.x;
  const int lane = t & 31;
  const int wave = t >> 5;
  const int wm   = wave & 3;                 // 32-row strip
  const int wn   = wave >> 2;                // (16*TN)-col strip
  const int row0 = blockIdx.x * BM;
  const int col0 = blockIdx.y * BN;

  const unsigned int smem_base =
      (unsigned int)(unsigned long long)(unsigned char*)smem;
  const unsigned long long gA =
      (unsigned long long)(const void*)(A + (size_t)row0 * lda);
  const unsigned long long gB =
      (unsigned long long)(const void*)(Bt + (size_t)col0 * ldb);

  // Copy slot: 8 bf16 (16 B) per thread per instr; 32 rows x 64 cols per instr.
  const int rC = t >> 3;                     // 0..31
  const int cC = (t & 7) * 8;                // 0..56

  v8f acc[2][TN];
#pragma unroll
  for (int i = 0; i < 2; ++i)
#pragma unroll
    for (int j = 0; j < TN; ++j) acc[i][j] = (v8f){};

  auto issue = [&](int buf, int k0) {
#pragma unroll
    for (int j = 0; j < NCPYA; ++j) {
      const int r = rC + j * 32;
      async_ld_b128(smem_base + buf * STAGEb + (unsigned)(r * BKS + cC) * 2,
                    gA, (unsigned)(r * lda + k0 + cC) * 2);
    }
#pragma unroll
    for (int j = 0; j < NCPYB; ++j) {
      const int r = rC + j * 32;
      async_ld_b128(smem_base + buf * STAGEb + ASZb + (unsigned)(r * BKS + cC) * 2,
                    gB, (unsigned)(r * ldb + k0 + cC) * 2);
    }
  };

  const int stages = K / BKS;
  issue(0, 0);

  const int lr = lane & 15;
  const int hf = lane >> 4;

  for (int s = 0; s < stages; ++s) {
    const int buf = s & 1;
    if (s + 1 < stages) {
      issue(buf ^ 1, (s + 1) * BKS);
      asm volatile("s_wait_asynccnt %0" :: "i"(NASYNC) : "memory");
    } else {
      asm volatile("s_wait_asynccnt 0" ::: "memory");
    }
    __syncthreads();

    const unsigned int* As = (const unsigned int*)(smem + buf * STAGEb);
    const unsigned int* Bs = (const unsigned int*)(smem + buf * STAGEb + ASZb);

#pragma unroll
    for (int ks = 0; ks < BKS / 32; ++ks) {
      v16bf a[2], b[TN];
#pragma unroll
      for (int i = 0; i < 2; ++i)
        a[i] = load_frag_a(As + (wm * 32 + i * 16 + lr) * (BKS / 2) + ks * 16, hf);
#pragma unroll
      for (int j = 0; j < TN; ++j)
        b[j] = load_frag_b(Bs + (wn * (16 * TN) + j * 16 + lr) * (BKS / 2) + ks * 16, hf);
#pragma unroll
      for (int i = 0; i < 2; ++i)
#pragma unroll
        for (int j = 0; j < TN; ++j)
          acc[i][j] = WMMA_BF16(a[i], b[j], acc[i][j]);
    }
    __syncthreads();
  }

  // Epilogue. C/D layout (ISA §7.12.2): N = lane%16, VGPR i -> M = i + 8*(lane/16).
#pragma unroll
  for (int tm = 0; tm < 2; ++tm) {
#pragma unroll
    for (int tn = 0; tn < TN; ++tn) {
      const int n     = col0 + wn * (16 * TN) + tn * 16 + lr;
      const int mbase = row0 + wm * 32 + tm * 16 + hf * 8;
      const float bv  = bias ? bias[n] : 0.0f;
      float vals[8];
#pragma unroll
      for (int i = 0; i < 8; ++i) {
        float v = acc[tm][tn][i] + bv;
        if (flags & 1) v = fmaxf(v, 0.0f);
        if (flags & 2) v = 1.0f / (1.0f + __expf(-v));
        vals[i] = v;
      }
      if (outF) {  // write-once 256MB stream: nontemporal, keep L2 for adj
#pragma unroll
        for (int i = 0; i < 8; ++i)
          __builtin_nontemporal_store(vals[i], &outF[(size_t)(mbase + i) * ldf + n]);
      }
      if (outB) {
#pragma unroll
        for (int i = 0; i < 8; ++i)
          outB[(size_t)(mbase + i) * ldob + n] = (__bf16)vals[i];
      }
      if (outT) {  // transposed bf16: 8 contiguous M per lane -> one b128 store
        union { uint4 q; __bf16 h[8]; } u;
#pragma unroll
        for (int i = 0; i < 8; ++i) u.h[i] = (__bf16)vals[i];
        *(uint4*)((__bf16*)outT + (size_t)n * ldt + mbase) = u.q;
      }
    }
  }
}

// f32 -> bf16, x4 vectorized, nontemporal reads (one-shot streams)
__global__ void __launch_bounds__(256)
gcn_cvt_bf16(const f32x4* __restrict__ in, u16x4* __restrict__ out, int n4) {
  int i = blockIdx.x * blockDim.x + threadIdx.x;
  if (i < n4) {
    f32x4 f = __builtin_nontemporal_load(&in[i]);
    u16x4 o;
    o.x = f2bf_bits(f.x); o.y = f2bf_bits(f.y);
    o.z = f2bf_bits(f.z); o.w = f2bf_bits(f.w);
    out[i] = o;
  }
}

// f32 [rows][cols] -> bf16 [cols][rows] (small weight matrices only)
__global__ void __launch_bounds__(256)
gcn_cvt_tr_bf16(const float* __restrict__ in, __bf16* __restrict__ out,
                int rows, int cols) {
  int i = blockIdx.x * blockDim.x + threadIdx.x;
  if (i < rows * cols) {
    int r = i / cols, c = i % cols;
    out[(size_t)c * rows + r] = (__bf16)in[i];
  }
}

extern "C" void kernel_launch(void* const* d_in, const int* in_sizes, int n_in,
                              void* d_out, int out_size, void* d_ws, size_t ws_size,
                              hipStream_t stream) {
  constexpr int N = 8192, NFEAT = 512, NHID = 256, NCLASS = 64;

  const float* x   = (const float*)d_in[0];
  const float* adj = (const float*)d_in[1];
  const float* W1  = (const float*)d_in[2];
  const float* b1  = (const float*)d_in[3];
  const float* W2  = (const float*)d_in[4];
  const float* b2  = (const float*)d_in[5];
  float* out = (float*)d_out;

  char* ws = (char*)d_ws;
  size_t off = 0;
  auto alloc = [&](size_t bytes) { char* p = ws + off; off += (bytes + 255) & ~(size_t)255; return p; };
  __bf16* adjb = (__bf16*)alloc((size_t)N * N * 2);         // 128 MB (fits L2)
  __bf16* xb   = (__bf16*)alloc((size_t)N * NFEAT * 2);     // 8 MB
  __bf16* W1t  = (__bf16*)alloc((size_t)NHID * NFEAT * 2);  // [NHID][NFEAT]
  __bf16* W2t  = (__bf16*)alloc((size_t)NCLASS * NHID * 2); // [NCLASS][NHID]
  __bf16* T1t  = (__bf16*)alloc((size_t)NHID * N * 2);      // (x@W1)^T
  __bf16* H    = (__bf16*)alloc((size_t)N * NHID * 2);      // [N][NHID]
  __bf16* T2t  = (__bf16*)alloc((size_t)NCLASS * N * 2);    // (H@W2)^T
  __bf16* Z    = (__bf16*)alloc((size_t)N * NCLASS * 2);    // [N][NCLASS]
  (void)ws_size; (void)in_sizes; (void)n_in; (void)out_size;

  dim3 blk(256);

  // Precision conversions
  int n_adj4 = N * N / 4;
  gcn_cvt_bf16<<<dim3(n_adj4 / 256), blk, 0, stream>>>((const f32x4*)adj, (u16x4*)adjb, n_adj4);
  int n_x4 = N * NFEAT / 4;
  gcn_cvt_bf16<<<dim3(n_x4 / 256), blk, 0, stream>>>((const f32x4*)x, (u16x4*)xb, n_x4);
  gcn_cvt_tr_bf16<<<dim3((NFEAT * NHID + 255) / 256), blk, 0, stream>>>(W1, W1t, NFEAT, NHID);
  gcn_cvt_tr_bf16<<<dim3((NHID * NCLASS + 255) / 256), blk, 0, stream>>>(W2, W2t, NHID, NCLASS);

  // 1) T1t = (x @ W1)^T : M=8192, N=NHID(256), K=NFEAT(512)
  gcn_wmma_gemm_bf16<4><<<dim3(N / BM, NHID / 128), blk, 0, stream>>>(
      xb, NFEAT, W1t, NFEAT, nullptr,
      nullptr, 0, nullptr, 0, T1t, N, NFEAT, 0);

  // 2) H = relu(adj @ T1 + b1) : K=8192
  gcn_wmma_gemm_bf16<4><<<dim3(N / BM, NHID / 128), blk, 0, stream>>>(
      adjb, N, T1t, N, b1,
      nullptr, 0, H, NHID, nullptr, 0, N, /*relu*/1);

  // 3) T2t = (H @ W2)^T : N=NCLASS(64), K=NHID(256)
  gcn_wmma_gemm_bf16<2><<<dim3(N / BM, NCLASS / 64), blk, 0, stream>>>(
      H, NHID, W2t, NHID, nullptr,
      nullptr, 0, nullptr, 0, T2t, N, NHID, 0);

  // 4) Z = adj @ T2 + b2 : K=8192
  gcn_wmma_gemm_bf16<2><<<dim3(N / BM, NCLASS / 64), blk, 0, stream>>>(
      adjb, N, T2t, N, b2,
      nullptr, 0, Z, NCLASS, nullptr, 0, N, 0);

  // 5) out = sigmoid(Z @ Z^T) : B^T == Z itself; K=NCLASS(64)
  gcn_wmma_gemm_bf16<4><<<dim3(N / BM, N / 128), blk, 0, stream>>>(
      Z, NCLASS, Z, NCLASS, nullptr,
      out, N, nullptr, 0, nullptr, 0, NCLASS, /*sigmoid*/2);
}
